// GraphAttentionLayer_28406913695897
// MI455X (gfx1250) — matrix operand order repaired and checked by reading, not compiled, not run on hardware
//
#include <hip/hip_runtime.h>
#include <hip/hip_bf16.h>

#define N_NODES 8192
#define FDIM    256
#define GAT_ALPHA 0.2f
#define MASK_VAL (-9000000000000000.0f)

typedef __attribute__((ext_vector_type(16))) __bf16 v16bf;
typedef __attribute__((ext_vector_type(8)))  float  v8f;
typedef __attribute__((ext_vector_type(2)))  float  v2f;
typedef __attribute__((ext_vector_type(4)))  int    v4i;

#if __has_builtin(__builtin_amdgcn_global_load_async_to_lds_b128)
#define HAVE_ASYNC_LDS 1
#else
#define HAVE_ASYNC_LDS 0
#endif

__device__ __forceinline__ void wait_async_lds() {
#if HAVE_ASYNC_LDS
#if __has_builtin(__builtin_amdgcn_s_wait_asynccnt)
    __builtin_amdgcn_s_wait_asynccnt(0);
#else
    asm volatile("s_wait_asynccnt 0" ::: "memory");
#endif
#endif
}

// ---------------------------------------------------------------------------
// Kernel 1: Wh = h @ W  (fp32 WMMA 16x16x4, one wave per 16x16 output tile)
// ---------------------------------------------------------------------------
__global__ __launch_bounds__(32) void wh_gemm_kernel(const float* __restrict__ h,
                                                     const float* __restrict__ W,
                                                     float* __restrict__ Wh) {
    const int tile = blockIdx.x;
    const int ti = tile >> 4;
    const int tf = tile & 15;
    const int i0 = ti * 16, f0 = tf * 16;
    const int lane = threadIdx.x;
    const int half = (lane >> 4) & 1;
    const int l16  = lane & 15;

    v8f acc = {};
#if __has_builtin(__builtin_amdgcn_wmma_f32_16x16x4_f32)
    for (int k = 0; k < FDIM; k += 4) {
        const int ka = k + half * 2;
        v2f a, b;
        a.x = h[(size_t)(i0 + l16) * FDIM + ka];
        a.y = h[(size_t)(i0 + l16) * FDIM + ka + 1];
        b.x = W[(size_t)ka       * FDIM + f0 + l16];
        b.y = W[(size_t)(ka + 1) * FDIM + f0 + l16];
        acc = __builtin_amdgcn_wmma_f32_16x16x4_f32(false, a, false, b,
                                                    (short)0, acc, false, false);
    }
#else
    for (int r = 0; r < 8; ++r) {
        const int row = i0 + r + half * 8;
        const int col = f0 + l16;
        float s = 0.f;
        for (int k = 0; k < FDIM; ++k)
            s += h[(size_t)row * FDIM + k] * W[(size_t)k * FDIM + col];
        acc[r] = s;
    }
#endif
#pragma unroll
    for (int r = 0; r < 8; ++r)
        Wh[(size_t)(i0 + r + half * 8) * FDIM + f0 + l16] = acc[r];
}

// ---------------------------------------------------------------------------
// Kernel 2: e1[i] = Wh[i,:] . a[:256],  e2[i] = Wh[i,:] . a[256:]
// ---------------------------------------------------------------------------
__global__ __launch_bounds__(256) void attn_vec_kernel(const float* __restrict__ Wh,
                                                       const float* __restrict__ a,
                                                       float* __restrict__ e1,
                                                       float* __restrict__ e2) {
    const int row  = (blockIdx.x * blockDim.x + threadIdx.x) >> 5;
    const int lane = threadIdx.x & 31;
    if (row >= N_NODES) return;
    const float* r = Wh + (size_t)row * FDIM;
    float s1 = 0.f, s2 = 0.f;
#pragma unroll
    for (int t = 0; t < 8; ++t) {
        const int idx = lane * 8 + t;
        const float v = r[idx];
        s1 += v * a[idx];
        s2 += v * a[FDIM + idx];
    }
#pragma unroll
    for (int off = 16; off; off >>= 1) {
        s1 += __shfl_xor(s1, off, 32);
        s2 += __shfl_xor(s2, off, 32);
    }
    if (lane == 0) { e1[row] = s1; e2[row] = s2; }
}

// ---------------------------------------------------------------------------
// Kernel 3: repack Wh (fp32) into bf16 WMMA-B-ready 32(K)x16(N) tiles.
// Lane: N = f0 + lane%16, K = 8h+t (t<8), 16+8h+(t-8) (t>=8), h=lane/16 -> one v16bf.
// ---------------------------------------------------------------------------
__global__ __launch_bounds__(256) void pack_b_kernel(const float* __restrict__ Wh,
                                                     v16bf* __restrict__ Bp) {
    const int tile = (blockIdx.x * blockDim.x + threadIdx.x) >> 5;
    const int lane = threadIdx.x & 31;
    if (tile >= (N_NODES / 32) * (FDIM / 16)) return;
    const int jt = tile >> 4;
    const int ft = tile & 15;
    const int j0 = jt * 32, f0 = ft * 16;
    const int n  = f0 + (lane & 15);
    const int Kb = (lane & 16) ? 16 : 0;
    v16bf v;
#pragma unroll
    for (int t = 0; t < 16; ++t)
        v[t] = (__bf16)Wh[(size_t)(j0 + (Kb >> 1) + (t < 8 ? t : 8 + t)) * FDIM + n];
    Bp[(size_t)tile * 32 + lane] = v;
}

// ---------------------------------------------------------------------------
// Kernel 4: split-K flash GAT. 512 blocks x 128 threads (4 waves).
// Block owns 16 rows. Wave (jhalf = wave&1, fhalf = wave>>1) owns a disjoint
// 32-col j-chunk and 8 f-tiles; per-wave online softmax; LDS epilogue merges
// the two j-halves. adj streams via async-to-LDS double buffering.
// ---------------------------------------------------------------------------
__global__ __launch_bounds__(128) void gat_flash_kernel(const int* __restrict__ adj,
                                                        const float* __restrict__ e1v,
                                                        const float* __restrict__ e2v,
                                                        const v16bf* __restrict__ Bp,
                                                        float* __restrict__ out) {
    __shared__ int   adjt[2][16 * 64];          // 8 KB double-buffered adj tile
    __shared__ float mstat[2][16], lstat[2][16];
    __shared__ float xch[2][8][8][32];          // 16 KB acc exchange (fhalf,t,r,lane)

    const int i0    = blockIdx.x * 16;
    const int tid   = threadIdx.x;
    const int wave  = tid >> 5;
    const int lane  = tid & 31;
    const int half  = lane >> 4;
    const int l16   = lane & 15;
    const int jhalf = wave & 1;
    const int fhalf = wave >> 1;

    const float e1 = e1v[i0 + l16];
    float m_run = MASK_VAL;
    float l_run = 0.f;
    v8f acc[8] = {v8f{}, v8f{}, v8f{}, v8f{}, v8f{}, v8f{}, v8f{}, v8f{}};

    const int base0 = half ? 8  : 0;            // A-matrix K group 1 (VGPRs 0-3)
    const int base1 = half ? 24 : 16;           // A-matrix K group 2 (VGPRs 4-7)
    const int sb    = half ? 8  : 0;            // acc VGPR r -> row sb + r

    auto stage = [&](int b, int jt) {           // stage 16x64 adj tile into LDS
        const int r = tid >> 3;                 // 0..15
        const int c = (tid & 7) * 8;            // 0,8,...,56
        const int* gsrc = adj + (size_t)(i0 + r) * N_NODES + jt * 64 + c;
        int* ldst = &adjt[b][r * 64 + c];
#if HAVE_ASYNC_LDS
        __builtin_amdgcn_global_load_async_to_lds_b128(
            (__attribute__((address_space(1))) v4i*)gsrc,
            (__attribute__((address_space(3))) v4i*)ldst, 0, 0);
        __builtin_amdgcn_global_load_async_to_lds_b128(
            (__attribute__((address_space(1))) v4i*)(gsrc + 4),
            (__attribute__((address_space(3))) v4i*)(ldst + 4), 0, 0);
#else
        *(int4*)ldst       = *(const int4*)gsrc;
        *(int4*)(ldst + 4) = *(const int4*)(gsrc + 4);
#endif
    };

    stage(0, 0);
    for (int jt = 0; jt < N_NODES / 64; ++jt) {
        const int buf = jt & 1;
        wait_async_lds();
        __syncthreads();                        // tile jt visible; prev buf free
        if (jt + 1 < N_NODES / 64) stage(buf ^ 1, jt + 1);

        const int j0 = jt * 64 + jhalf * 32;
        const int* arow = &adjt[buf][l16 * 64 + jhalf * 32];

        float s[16];
        float tmax = MASK_VAL;
#pragma unroll
        for (int t = 0; t < 16; ++t) {
            const int K = (t < 8) ? (base0 + t) : (base1 + t - 8);
            const float raw = e1 + e2v[j0 + K];
            const float lr  = raw > 0.f ? raw : GAT_ALPHA * raw;
            const float v   = (arow[K] > 0) ? lr : MASK_VAL;
            s[t] = v;
            tmax = fmaxf(tmax, v);
        }
        tmax = fmaxf(tmax, __shfl_xor(tmax, 16, 32));
        const float m_new = fmaxf(m_run, tmax);
        const float scale = __expf(m_run - m_new);
        if (__ballot(m_new > m_run)) {          // wave-uniform
#pragma unroll
            for (int r = 0; r < 8; ++r) {
                const float sc = __shfl(scale, sb + r, 32);
#pragma unroll
                for (int t = 0; t < 8; ++t) acc[t][r] *= sc;
            }
        }
        l_run *= scale;
        m_run  = m_new;

        v16bf aP;
        float psum = 0.f;
#pragma unroll
        for (int t = 0; t < 16; ++t) {
            const float p = __expf(s[t] - m_new);
            psum += p;
            aP[t] = (__bf16)p;
        }
        psum  += __shfl_xor(psum, 16, 32);
        l_run += psum;

        const int jdx = jt * 2 + jhalf;         // global 32-col tile index
#pragma unroll
        for (int t = 0; t < 8; ++t) {
            const int ft = fhalf * 8 + t;
            const v16bf bT = Bp[(size_t)(jdx * 16 + ft) * 32 + lane];
            acc[t] = __builtin_amdgcn_wmma_f32_16x16x32_bf16(
                false, aP, false, bT, (short)0, acc[t], false, false);
        }
    }

    // ---- merge the two j-halves and finalize -----------------------------
    if (fhalf == 0 && lane < 16) {
        mstat[jhalf][l16] = m_run;
        lstat[jhalf][l16] = l_run;
    }
    __syncthreads();
    const float m0 = mstat[0][l16], m1 = mstat[1][l16];
    const float l0 = lstat[0][l16], l1 = lstat[1][l16];
    const float M  = fmaxf(m0, m1);
    const float L  = l0 * __expf(m0 - M) + l1 * __expf(m1 - M);
    const float sw_row  = __expf((jhalf ? m1 : m0) - M);   // scale for own acc
    const float inv_row = 1.0f / L;

    if (jhalf == 1) {
#pragma unroll
        for (int r = 0; r < 8; ++r) {
            const float sr = __shfl(sw_row, sb + r, 32);
#pragma unroll
            for (int t = 0; t < 8; ++t)
                xch[fhalf][t][r][lane] = acc[t][r] * sr;
        }
    }
    __syncthreads();
    if (jhalf == 0) {
#pragma unroll
        for (int r = 0; r < 8; ++r) {
            const float sr = __shfl(sw_row,  sb + r, 32);
            const float iv = __shfl(inv_row, sb + r, 32);
#pragma unroll
            for (int t = 0; t < 8; ++t) {
                const int ft = fhalf * 8 + t;
                const float hp = (acc[t][r] * sr + xch[fhalf][t][r][lane]) * iv;
                const float o  = hp > 0.f ? hp : (__expf(hp) - 1.f);
                out[(size_t)(i0 + sb + r) * FDIM + ft * 16 + l16] = o;
            }
        }
    }
}

// ---------------------------------------------------------------------------
extern "C" void kernel_launch(void* const* d_in, const int* in_sizes, int n_in,
                              void* d_out, int out_size, void* d_ws, size_t ws_size,
                              hipStream_t stream) {
    (void)in_sizes; (void)n_in; (void)out_size; (void)ws_size;
    const float* h   = (const float*)d_in[0];   // [8192, 256]
    const int*   adj = (const int*)  d_in[1];   // [8192, 8192]
    const float* W   = (const float*)d_in[2];   // [256, 256]
    const float* a   = (const float*)d_in[3];   // [512, 1]
    float* out = (float*)d_out;                 // [8192, 256]

    char* ws = (char*)d_ws;
    float* Wh = (float*)ws;                                   // 8 MB
    v16bf* Bp = (v16bf*)(ws + (size_t)N_NODES * FDIM * 4);    // 4 MB
    float* e1 = (float*)(ws + (size_t)N_NODES * FDIM * 6);    // 32 KB
    float* e2 = e1 + N_NODES;                                 // 32 KB

    wh_gemm_kernel<<<(N_NODES / 16) * (FDIM / 16), 32, 0, stream>>>(h, W, Wh);
    attn_vec_kernel<<<(N_NODES * 32) / 256, 256, 0, stream>>>(Wh, a, e1, e2);
    pack_b_kernel<<<((N_NODES / 32) * (FDIM / 16) * 32) / 256, 256, 0, stream>>>(Wh, Bp);
    gat_flash_kernel<<<N_NODES / 16, 128, 0, stream>>>(adj, e1, e2, Bp, out);
}
